// TopkSAE_61452392071745
// MI455X (gfx1250) — compile-verified, matrix operand order checked by reading, not run on hardware
//
#include <hip/hip_runtime.h>

#define HIDDEN 4096
#define LATENT 32768
#define NTOK   4096
#define TOPK   64

#define BM 256
#define BN 128
#define BK 32
#define BKP 40   // padded LDS row stride (conflict-free b128 reads)

typedef __bf16 v16bf  __attribute__((ext_vector_type(16)));
typedef __bf16 bf16x8 __attribute__((ext_vector_type(8)));
typedef __bf16 bf16x4 __attribute__((ext_vector_type(4)));
typedef float  v8f    __attribute__((ext_vector_type(8)));
typedef int    v4i    __attribute__((ext_vector_type(4)));

#define AS1 __attribute__((address_space(1)))
#define AS3 __attribute__((address_space(3)))

#if __has_builtin(__builtin_amdgcn_global_load_async_to_lds_b128) && \
    __has_builtin(__builtin_amdgcn_s_wait_asynccnt)
#define SAE_ASYNC_LDS 1
#endif

static __device__ __forceinline__ v8f wmma_bf16(v16bf a, v16bf b, v8f c) {
  return __builtin_amdgcn_wmma_f32_16x16x32_bf16(false, a, false, b, (short)0, c,
                                                 false, false);
}

// Build a 16x16x32 bf16 A/B fragment from LDS: lanes 0-15 take K {g..g+7, g+16..g+23}
static __device__ __forceinline__ v16bf make_frag(const __bf16* p) {
  bf16x8 x0 = *(const bf16x8*)(p);
  bf16x8 x1 = *(const bf16x8*)(p + 16);
  v16bf r;
#pragma unroll
  for (int i = 0; i < 8; ++i) { r[i] = x0[i]; r[8 + i] = x1[i]; }
  return r;
}

// latents[NTOK,LATENT] = (x - pre_bias)[NTOK,HIDDEN] * enc_w[LATENT,HIDDEN]^T + latent_bias
// fp32 in/out; bf16 hi/lo split: 3 WMMAs (hi*hi + lo*hi + hi*lo) per 16x16x32 step.
__global__ __launch_bounds__(256) void sae_enc_gemm(
    const float* __restrict__ A, const float* __restrict__ B,
    float* __restrict__ C,
    const float* __restrict__ subk,   // pre_bias, len HIDDEN
    const float* __restrict__ addn)   // latent_bias, len LATENT
{
  constexpr int M = NTOK, N = LATENT, K = HIDDEN;
  const int m0 = blockIdx.x * BM;     // m fastest-varying -> enc_w tile L2 reuse
  const int n0 = blockIdx.y * BN;
  const int tid  = threadIdx.x;
  const int lane = tid & 31;
  const int wav  = tid >> 5;
  const int wm = wav >> 1;            // 0..3 -> 64 rows each
  const int wn = wav & 1;             // 0..1 -> 64 cols each
  const int lr = lane & 15;
  const int g  = (lane < 16) ? 0 : 8;

  __shared__ __bf16 sA[2][BM][BKP];   // [hi/lo][row][k]  40KB
  __shared__ __bf16 sB[2][BN][BKP];   //                  20KB

  v8f acc[4][4];
  const v8f vzero = {0.f, 0.f, 0.f, 0.f, 0.f, 0.f, 0.f, 0.f};
#pragma unroll
  for (int mt = 0; mt < 4; ++mt)
#pragma unroll
    for (int nt = 0; nt < 4; ++nt) acc[mt][nt] = vzero;

  const int tcol = (tid & 7) << 2;    // invariant per thread (256 % 8 == 0)
  const int trow = tid >> 3;          // 0..31

  for (int kt = 0; kt < K; kt += BK) {
    // ---- stage fp32 tiles -> bf16 hi/lo in LDS ----
    const float4 sb4 = *reinterpret_cast<const float4*>(subk + kt + tcol);
    const float sb[4] = {sb4.x, sb4.y, sb4.z, sb4.w};

    if (kt + BK < K) {  // prefetch next k-stage rows (global_prefetch_b8)
      __builtin_prefetch(A + (size_t)(m0 + trow) * K + kt + BK + tcol, 0, 1);
      __builtin_prefetch(B + (size_t)(n0 + trow) * K + kt + BK + tcol, 0, 1);
    }

#pragma unroll
    for (int i = 0; i < BM / 32; ++i) {           // 8 chunks of A
      int row = trow + i * 32;
      const float4 va = *reinterpret_cast<const float4*>(
          A + (size_t)(m0 + row) * K + kt + tcol);
      float fa[4] = {va.x, va.y, va.z, va.w};
      bf16x4 h4, l4;
#pragma unroll
      for (int j = 0; j < 4; ++j) {
        float xv = fa[j] - sb[j];
        __bf16 h = (__bf16)xv;
        h4[j] = h; l4[j] = (__bf16)(xv - (float)h);
      }
      *(bf16x4*)&sA[0][row][tcol] = h4;
      *(bf16x4*)&sA[1][row][tcol] = l4;
    }
#pragma unroll
    for (int i = 0; i < BN / 32; ++i) {           // 4 chunks of B
      int row = trow + i * 32;
      const float4 vb = *reinterpret_cast<const float4*>(
          B + (size_t)(n0 + row) * K + kt + tcol);
      float fb[4] = {vb.x, vb.y, vb.z, vb.w};
      bf16x4 h4, l4;
#pragma unroll
      for (int j = 0; j < 4; ++j) {
        __bf16 h = (__bf16)fb[j];
        h4[j] = h; l4[j] = (__bf16)(fb[j] - (float)h);
      }
      *(bf16x4*)&sB[0][row][tcol] = h4;
      *(bf16x4*)&sB[1][row][tcol] = l4;
    }
    __syncthreads();

    // ---- WMMA: 4x4 C-tiles, 3 wmma each = 48 per k-stage ----
    v16bf bh[4], bl[4];
#pragma unroll
    for (int nt = 0; nt < 4; ++nt) {
      int br = wn * 64 + nt * 16 + lr;
      bh[nt] = make_frag(&sB[0][br][g]);
      bl[nt] = make_frag(&sB[1][br][g]);
    }
#pragma unroll
    for (int mt = 0; mt < 4; ++mt) {
      int ar = wm * 64 + mt * 16 + lr;
      v16bf ah = make_frag(&sA[0][ar][g]);
      v16bf al = make_frag(&sA[1][ar][g]);
#pragma unroll
      for (int nt = 0; nt < 4; ++nt) {
        acc[mt][nt] = wmma_bf16(ah, bh[nt], acc[mt][nt]);
        acc[mt][nt] = wmma_bf16(al, bh[nt], acc[mt][nt]);
        acc[mt][nt] = wmma_bf16(ah, bl[nt], acc[mt][nt]);
      }
    }
    __syncthreads();
  }

  // ---- epilogue: C/D layout: vgpr v -> M=v (lanes<16) / v+8; N = lane&15 ----
#pragma unroll
  for (int mt = 0; mt < 4; ++mt) {
    int rbase = m0 + wm * 64 + mt * 16 + ((lane < 16) ? 0 : 8);
#pragma unroll
    for (int nt = 0; nt < 4; ++nt) {
      int col = n0 + wn * 64 + nt * 16 + lr;
      float bias = addn[col];
#pragma unroll
      for (int v = 0; v < 8; ++v) {
        C[(size_t)(rbase + v) * N + col] = acc[mt][nt][v] + bias;
      }
    }
  }
}

static __device__ __forceinline__ unsigned order_key(float f) {
  unsigned u = __float_as_uint(f);
  return (u & 0x80000000u) ? ~u : (u | 0x80000000u);  // monotone float order
}

// Per-token 4-round radix select of top-64; filter row in place; emit
// deterministic, index-ordered compacted (val, idx) transposed lists.
__global__ __launch_bounds__(256) void topk_select(
    float* __restrict__ lat,      // [NTOK, LATENT] in/out
    float* __restrict__ valsT,    // [TOPK, NTOK]
    int*   __restrict__ idxT)     // [TOPK, NTOK]
{
  const int t = blockIdx.x;
  float* p = lat + (size_t)t * LATENT;
  __shared__ unsigned hist[256];
  __shared__ unsigned s_prefix;
  __shared__ int s_need;
  __shared__ unsigned gcnt[256], ecnt[256], gscan[256], escan[256];

  unsigned prefix = 0;
  int need = TOPK;

  for (int round = 0; round < 4; ++round) {
    const int shift = 24 - 8 * round;
    hist[threadIdx.x] = 0;
    __syncthreads();
    for (int i = threadIdx.x; i < LATENT; i += 256) {
      unsigned key = order_key(p[i]);
      bool match = (round == 0) ||
                   ((key >> (shift + 8)) == (prefix >> (shift + 8)));
      if (match) atomicAdd(&hist[(key >> shift) & 255u], 1u);
    }
    __syncthreads();
    if (threadIdx.x == 0) {
      int cum = 0, b = 256;
      while (b > 0) { --b; cum += (int)hist[b]; if (cum >= need) break; }
      s_need = need - (cum - (int)hist[b]);
      s_prefix = prefix | ((unsigned)b << shift);
    }
    __syncthreads();
    prefix = s_prefix;
    need = s_need;
  }

  const unsigned T = prefix;        // exact key of the 64th-largest element
  const int keepEq = need;          // how many ties at T to keep (index order)

  // deterministic compaction over contiguous per-thread chunks
  const int span = LATENT / 256;
  const int i0 = threadIdx.x * span;
  unsigned cg = 0, ce = 0;
  for (int i = i0; i < i0 + span; ++i) {
    unsigned key = order_key(p[i]);
    cg += (key > T);
    ce += (key == T);
  }
  gcnt[threadIdx.x] = cg; ecnt[threadIdx.x] = ce;
  __syncthreads();
  if (threadIdx.x == 0) {
    unsigned ag = 0, ae = 0;
    for (int j = 0; j < 256; ++j) {
      gscan[j] = ag; escan[j] = ae;
      ag += gcnt[j]; ae += ecnt[j];
    }
  }
  __syncthreads();

  unsigned lg = gscan[threadIdx.x], le = escan[threadIdx.x];
  for (int i = i0; i < i0 + span; ++i) {
    float f = p[i];
    unsigned key = order_key(f);
    bool keep = false;
    unsigned slot = 0;
    if (key > T) {
      keep = true;
      slot = lg + ((le < (unsigned)keepEq) ? le : (unsigned)keepEq);
      ++lg;
    } else if (key == T) {
      if (le < (unsigned)keepEq) { keep = true; slot = lg + le; }
      ++le;
    }
    if (keep) { valsT[slot * NTOK + t] = f; idxT[slot * NTOK + t] = i; }
    p[i] = keep ? f : 0.0f;
  }
}

// Sparse decoder: one block per hidden row h; stage contiguous 128KB dec_w row
// in LDS via CDNA5 async global->LDS copies when available, then 64 LDS
// gathers per token. dec_w streams from HBM exactly once.
__global__ __launch_bounds__(256) void sae_decode(
    const float* __restrict__ dec_w,   // [HIDDEN, LATENT]
    const float* __restrict__ valsT,   // [TOPK, NTOK]
    const int*   __restrict__ idxT,    // [TOPK, NTOK]
    const float* __restrict__ pre_bias,
    float* __restrict__ xhat)          // [NTOK, HIDDEN]
{
  const int h = blockIdx.x;
  extern __shared__ float row[];       // LATENT floats = 128KB
  const float* src = dec_w + (size_t)h * LATENT;

#if defined(SAE_ASYNC_LDS)
  for (int i = threadIdx.x; i < LATENT / 4; i += 256) {
    __builtin_amdgcn_global_load_async_to_lds_b128(
        (AS1 v4i*)(AS1 void*)(src + (size_t)i * 4),
        (AS3 v4i*)(AS3 void*)(row + (size_t)i * 4),
        0, 0);
  }
  __builtin_amdgcn_s_wait_asynccnt(0);
  __syncthreads();
#else
  const float4* s4 = reinterpret_cast<const float4*>(src);
  float4* d4 = reinterpret_cast<float4*>(row);
  for (int i = threadIdx.x; i < LATENT / 4; i += 256) d4[i] = s4[i];
  __syncthreads();
#endif

  const float pb = pre_bias[h];
  for (int t = threadIdx.x; t < NTOK; t += 256) {
    float s = pb;
#pragma unroll 8
    for (int k = 0; k < TOPK; ++k) {
      float v = valsT[k * NTOK + t];        // coalesced across lanes
      int  ix = idxT[k * NTOK + t];
      s = fmaf(v, row[ix], s);
    }
    xhat[(size_t)t * HIDDEN + h] = s;
  }
}

extern "C" void kernel_launch(void* const* d_in, const int* in_sizes, int n_in,
                              void* d_out, int out_size, void* d_ws, size_t ws_size,
                              hipStream_t stream) {
  (void)in_sizes; (void)n_in; (void)out_size; (void)ws_size;
  const float* x   = (const float*)d_in[0];
  const float* enc = (const float*)d_in[1];   // [LATENT, HIDDEN]
  const float* dec = (const float*)d_in[2];   // [HIDDEN, LATENT]
  const float* pb  = (const float*)d_in[3];   // [HIDDEN]
  const float* lb  = (const float*)d_in[4];   // [LATENT]

  float* latents = (float*)d_out;                              // [NTOK, LATENT]
  float* xhat    = latents + (size_t)NTOK * LATENT;            // [NTOK, HIDDEN]

  float* valsT = (float*)d_ws;                                 // [TOPK, NTOK] 1MB
  int*   idxT  = (int*)((char*)d_ws + (size_t)TOPK * NTOK * sizeof(float)); // 1MB

  dim3 blk(256);

  // Encoder: pre_acts = (x - pre_bias) @ enc_w^T + latent_bias -> latents region
  dim3 g1(NTOK / BM, LATENT / BN);
  sae_enc_gemm<<<g1, blk, 0, stream>>>(x, enc, latents, pb, lb);

  // Top-64 per token: filter latents in place + compact (val, idx) lists
  topk_select<<<dim3(NTOK), blk, 0, stream>>>(latents, valsT, idxT);

  // Sparse decoder: x_hat = latents @ dec_w^T + pre_bias
  sae_decode<<<dim3(HIDDEN), blk, LATENT * sizeof(float), stream>>>(
      dec, valsT, idxT, pb, xhat);
}
